// MultiHeadSelfAttention_34746285424933
// MI455X (gfx1250) — matrix operand order repaired and checked by reading, not compiled, not run on hardware
//
#include <hip/hip_runtime.h>
#include <cstdint>

typedef __bf16 v16bf __attribute__((ext_vector_type(16)));
typedef float  v8f   __attribute__((ext_vector_type(8)));

static __device__ __forceinline__ unsigned short f2bf(float f) {
  union { float f; unsigned u; } x; x.f = f;
  unsigned r = x.u + 0x7fffu + ((x.u >> 16) & 1u);   // round-to-nearest-even
  return (unsigned short)(r >> 16);
}

// Build a 16-element bf16 fragment from two aligned 16-byte chunks.
static __device__ __forceinline__ v16bf ld_frag(const unsigned short* p0,
                                                const unsigned short* p1) {
  union { uint4 q[2]; v16bf v; } u;
  u.q[0] = *reinterpret_cast<const uint4*>(p0);
  u.q[1] = *reinterpret_cast<const uint4*>(p1);
  return u.v;
}

// Async global->LDS 16B copy (per active lane). LDS offset = low 32 bits of the
// generic pointer.
static __device__ __forceinline__ void async_copy_b128(const unsigned short* gsrc,
                                                       void* lds_dst) {
  unsigned loff = (unsigned)(uintptr_t)lds_dst;
  asm volatile("global_load_async_to_lds_b128 %0, %1, off"
               :: "v"(loff), "v"(gsrc) : "memory");
}
static __device__ __forceinline__ void async_wait0() {
  asm volatile("s_wait_asynccnt 0x0" ::: "memory");
}

// ---------------------------------------------------------------------------
// Generic bf16 GEMM:  C(f32)[M,N] = A(bf16)[M,K] * B(bf16)[K,N], row-major.
// Block tile 128x128, 8 waves (4x2), wave tile 32x64 = 2x4 WMMA tiles.
// 2-stage software pipeline: async-DMA A(i+1) + global B(i+1) issued before
// computing tile i; one barrier per K-step; DMA runs under the WMMA shadow.
// ---------------------------------------------------------------------------
__global__ __launch_bounds__(256) void gemm_bf16_f32(
    const unsigned short* __restrict__ A, const unsigned short* __restrict__ Bm,
    float* __restrict__ C, int M, int N, int K) {
  __shared__ unsigned short As[2][128][40];   // row stride 80B (16B aligned)
  __shared__ unsigned short Bs[2][128][40];   // stored transposed: Bs[s][n][k]

  const int tid  = threadIdx.x;
  const int lane = tid & 31, wave = tid >> 5;
  const int hi = (lane >> 4) & 1, lo = lane & 15;
  const int wm = (wave & 3) * 32, wn = (wave >> 2) * 64;
  const int m0 = blockIdx.y * 128, n0 = blockIdx.x * 128;

  v8f acc[2][4];
#pragma unroll
  for (int i = 0; i < 2; ++i)
#pragma unroll
    for (int j = 0; j < 4; ++j)
#pragma unroll
      for (int r = 0; r < 8; ++r) acc[i][j][r] = 0.0f;

  const int ar = tid >> 1;            // A row in tile (0..127)
  const int ac = (tid & 1) * 16;      // A col base (16 halves per thread)
  const int bk = tid >> 3;            // B k row (0..31)
  const int bn = (tid & 7) * 16;      // B n base (16 halves per thread)

  const unsigned short* gA = A + (size_t)(m0 + ar) * K + ac;
  const unsigned short* gB = Bm + (size_t)bk * N + n0 + bn;

  const int nk = K >> 5;

  // ---- prologue: stage tile 0 into buffer 0
  {
    async_copy_b128(gA,     &As[0][ar][ac]);
    async_copy_b128(gA + 8, &As[0][ar][ac + 8]);
    union { uint4 q; unsigned short s[8]; } b0, b1;
    b0.q = *(const uint4*)gB;
    b1.q = *(const uint4*)(gB + 8);
#pragma unroll
    for (int e = 0; e < 8; ++e) Bs[0][bn + e][bk] = b0.s[e];
#pragma unroll
    for (int e = 0; e < 8; ++e) Bs[0][bn + 8 + e][bk] = b1.s[e];
    async_wait0();
  }
  __syncthreads();

  for (int i = 0; i < nk; ++i) {
    const int cur = i & 1, nxt = cur ^ 1;
    const bool have_next = (i + 1) < nk;
    union { uint4 q; unsigned short s[8]; } nb0, nb1;
    if (have_next) {
      const int k1 = (i + 1) << 5;
      // kick off next tile's DMA + global loads before doing any math
      async_copy_b128(gA + k1,     &As[nxt][ar][ac]);
      async_copy_b128(gA + k1 + 8, &As[nxt][ar][ac + 8]);
      nb0.q = *(const uint4*)(gB + (size_t)k1 * N);
      nb1.q = *(const uint4*)(gB + (size_t)k1 * N + 8);
    }

    // ---- compute tile i
    v16bf af[2], bfr[4];
#pragma unroll
    for (int ii = 0; ii < 2; ++ii) {
      // A-frag (16x32): lanes 0-15 chunks {0..7,16..23}, lanes 16-31 {8..15,24..31}
      const unsigned short* p = &As[cur][wm + ii * 16 + lo][hi * 8];
      af[ii] = ld_frag(p, p + 16);
    }
#pragma unroll
    for (int j = 0; j < 4; ++j) {
      // B-frag (32x16): lane = column, K chunks {hi*8..+7, 16+hi*8..+7}
      const unsigned short* p = &Bs[cur][wn + j * 16 + lo][hi * 8];
      bfr[j] = ld_frag(p, p + 16);
    }
#pragma unroll
    for (int ii = 0; ii < 2; ++ii)
#pragma unroll
      for (int j = 0; j < 4; ++j)
        acc[ii][j] = __builtin_amdgcn_wmma_f32_16x16x32_bf16(
            false, af[ii], false, bfr[j], (short)0, acc[ii][j], false, false);

    if (have_next) {
#pragma unroll
      for (int e = 0; e < 8; ++e) Bs[nxt][bn + e][bk] = nb0.s[e];
#pragma unroll
      for (int e = 0; e < 8; ++e) Bs[nxt][bn + 8 + e][bk] = nb1.s[e];
      async_wait0();
    }
    __syncthreads();
  }

#pragma unroll
  for (int i = 0; i < 2; ++i)
#pragma unroll
    for (int j = 0; j < 4; ++j)
#pragma unroll
      for (int r = 0; r < 8; ++r) {
        int m = m0 + wm + i * 16 + r + 8 * hi;     // C/D layout: M = r + 8*laneHi
        int n = n0 + wn + j * 16 + lo;             // N = lane&15
        C[(size_t)m * N + n] = acc[i][j][r];
      }
}

// ---------------------------------------------------------------------------
// Pack Wq|Wk|Wv into one bf16 [1024 x 1536] matrix.
// ---------------------------------------------------------------------------
__global__ void pack_wqkv(const float* __restrict__ Wq, const float* __restrict__ Wk,
                          const float* __restrict__ Wv, unsigned short* __restrict__ out) {
  int i = blockIdx.x * blockDim.x + threadIdx.x;
  int stride = gridDim.x * blockDim.x;
  for (; i < 1024 * 1536; i += stride) {
    int k = i / 1536, n = i % 1536;
    float v;
    if (n < 1024)      v = Wq[(size_t)k * 1024 + n];
    else if (n < 1280) v = Wk[(size_t)k * 256 + (n - 1024)];
    else               v = Wv[(size_t)k * 256 + (n - 1280)];
    out[i] = f2bf(v);
  }
}

__global__ void cvt_f32_bf16(const float* __restrict__ in,
                             unsigned short* __restrict__ out, int n) {
  int i = blockIdx.x * blockDim.x + threadIdx.x;
  int stride = gridDim.x * blockDim.x;
  for (; i < n; i += stride) out[i] = f2bf(in[i]);
}

// ---------------------------------------------------------------------------
// RoPE + RMSNorm on q/k, plain convert for v. One wave per (b,t,head).
// Lane l holds the RoPE pair (d=l, d=l+32). roles: 0-15 q, 16-19 k, 20-23 v.
// ---------------------------------------------------------------------------
__global__ __launch_bounds__(128) void rope_rms(
    const float* __restrict__ qkv, const float* __restrict__ cosT,
    const float* __restrict__ sinT, unsigned short* __restrict__ qb,
    unsigned short* __restrict__ kb, unsigned short* __restrict__ vb) {
  const int lane = threadIdx.x & 31, wave = threadIdx.x >> 5;
  const int task = blockIdx.x * 4 + wave;
  const int bt = task / 24, role = task % 24;
  const int b = bt / 2048, t = bt % 2048;

  const float* src;
  unsigned short* dst;
  bool do_rope = true;
  if (role < 16) {
    src = qkv + (size_t)bt * 1536 + role * 64;
    dst = qb + ((size_t)(b * 16 + role) * 2048 + t) * 64;
  } else if (role < 20) {
    int kvh = role - 16;
    src = qkv + (size_t)bt * 1536 + 1024 + kvh * 64;
    dst = kb + ((size_t)(b * 4 + kvh) * 2048 + t) * 64;
  } else {
    int kvh = role - 20;
    src = qkv + (size_t)bt * 1536 + 1280 + kvh * 64;
    dst = vb + ((size_t)(b * 4 + kvh) * 2048 + t) * 64;
    do_rope = false;
  }
  float x1 = src[lane], x2 = src[lane + 32];
  if (do_rope) {
    float c = cosT[t * 32 + lane], s = sinT[t * 32 + lane];
    float y1 = x1 * c + x2 * s;
    float y2 = -x1 * s + x2 * c;
    float ssq = y1 * y1 + y2 * y2;
#pragma unroll
    for (int m = 16; m; m >>= 1) ssq += __shfl_xor(ssq, m, 32);
    float rn = rsqrtf(ssq * (1.0f / 64.0f) + 1e-6f);
    dst[lane]      = f2bf(y1 * rn);
    dst[lane + 32] = f2bf(y2 * rn);
  } else {
    dst[lane]      = f2bf(x1);
    dst[lane + 32] = f2bf(x2);
  }
}

// ---------------------------------------------------------------------------
// Causal flash attention. Grid: (T/64, B*H). 4 waves; wave owns 16 q rows.
// HD=64, key blocks of 64. 2-stage pipeline: next block's async K DMA and
// V global loads issued before this block's softmax; single barrier per block.
// ---------------------------------------------------------------------------
__global__ __launch_bounds__(128) void attn_fwd(
    const unsigned short* __restrict__ qb, const unsigned short* __restrict__ kb,
    const unsigned short* __restrict__ vb, unsigned short* __restrict__ yb) {
  __shared__ unsigned short Ks[2][64][72];     // [key][d], 144B rows
  __shared__ unsigned short Vt[2][64][72];     // [d][key] (transposed)
  __shared__ unsigned short Ps[4][16][72];     // per-wave P staging [row][key]

  const int tid = threadIdx.x, lane = tid & 31, wave = tid >> 5;
  const int hi = (lane >> 4) & 1, lo = lane & 15;
  const int qblk = blockIdx.x;
  const int b = blockIdx.y / 16, h = blockIdx.y % 16, kvh = h >> 2;
  const int tbase = qblk * 64 + wave * 16;

  const unsigned short* qhead = qb + (size_t)(b * 16 + h) * 2048 * 64;
  const unsigned short* khead = kb + (size_t)(b * 4 + kvh) * 2048 * 64;
  const unsigned short* vhead = vb + (size_t)(b * 4 + kvh) * 2048 * 64;

  // per-thread staging coords: 4 chunks of 16B (row = key, ch = d base)
  const int srow[4] = { (tid + 0)   >> 3, (tid + 128) >> 3,
                        (tid + 256) >> 3, (tid + 384) >> 3 };
  const int sch = (tid & 7) * 8;

  v16bf qa[2];
#pragma unroll
  for (int ks = 0; ks < 2; ++ks) {
    const unsigned short* p = qhead + (size_t)(tbase + lo) * 64 + hi * 8 + ks * 32;
    qa[ks] = ld_frag(p, p + 16);
  }

  float mrun[8], lrun[8];
#pragma unroll
  for (int r = 0; r < 8; ++r) { mrun[r] = -1e30f; lrun[r] = 0.0f; }
  v8f accO[4];
#pragma unroll
  for (int dt = 0; dt < 4; ++dt)
#pragma unroll
    for (int r = 0; r < 8; ++r) accO[dt][r] = 0.0f;

  // ---- prologue: stage key-block 0 into buffer 0
  {
#pragma unroll
    for (int it = 0; it < 4; ++it) {
      async_copy_b128(khead + (size_t)srow[it] * 64 + sch, &Ks[0][srow[it]][sch]);
      union { uint4 q; unsigned short s[8]; } vu;
      vu.q = *(const uint4*)(vhead + (size_t)srow[it] * 64 + sch);
#pragma unroll
      for (int e = 0; e < 8; ++e) Vt[0][sch + e][srow[it]] = vu.s[e];
    }
    async_wait0();
  }
  __syncthreads();

  for (int j = 0; j <= qblk; ++j) {
    const int cur = j & 1, nxt = cur ^ 1;
    const bool have_next = j < qblk;
    uint4 vreg[4];
    if (have_next) {
      const size_t base = (size_t)(j + 1) * 64 * 64;
#pragma unroll
      for (int it = 0; it < 4; ++it) {
        async_copy_b128(khead + base + (size_t)srow[it] * 64 + sch,
                        &Ks[nxt][srow[it]][sch]);
        vreg[it] = *(const uint4*)(vhead + base + (size_t)srow[it] * 64 + sch);
      }
    }

    // S = Q @ K^T  (4 N-tiles x 2 K-steps)
    v8f accS[4];
#pragma unroll
    for (int nt = 0; nt < 4; ++nt) {
#pragma unroll
      for (int r = 0; r < 8; ++r) accS[nt][r] = 0.0f;
#pragma unroll
      for (int ks = 0; ks < 2; ++ks) {
        const unsigned short* p = &Ks[cur][nt * 16 + lo][hi * 8 + ks * 32];
        v16bf bfr = ld_frag(p, p + 16);
        accS[nt] = __builtin_amdgcn_wmma_f32_16x16x32_bf16(
            false, qa[ks], false, bfr, (short)0, accS[nt], false, false);
      }
    }

    const float scale = 0.125f;                // 1/sqrt(64)
    const int qrow0 = tbase + 8 * hi;
    const int kcol0 = j * 64 + lo;
#pragma unroll
    for (int r = 0; r < 8; ++r) {
      float s[4];
#pragma unroll
      for (int nt = 0; nt < 4; ++nt) {
        float v = accS[nt][r] * scale;
        s[nt] = (kcol0 + nt * 16 <= qrow0 + r) ? v : -1e30f;   // causal mask
      }
      float rm = fmaxf(fmaxf(s[0], s[1]), fmaxf(s[2], s[3]));
#pragma unroll
      for (int msk = 1; msk < 16; msk <<= 1) rm = fmaxf(rm, __shfl_xor(rm, msk, 16));
      float mnew = fmaxf(mrun[r], rm);
      float corr = __expf(mrun[r] - mnew);
      float rs = 0.0f;
#pragma unroll
      for (int nt = 0; nt < 4; ++nt) { s[nt] = __expf(s[nt] - mnew); rs += s[nt]; }
#pragma unroll
      for (int msk = 1; msk < 16; msk <<= 1) rs += __shfl_xor(rs, msk, 16);
      lrun[r] = lrun[r] * corr + rs;
      mrun[r] = mnew;
#pragma unroll
      for (int dt = 0; dt < 4; ++dt) accO[dt][r] *= corr;
#pragma unroll
      for (int nt = 0; nt < 4; ++nt)
        Ps[wave][r + 8 * hi][nt * 16 + lo] = f2bf(s[nt]);  // C-layout -> [m][k]
    }

    // stage next V tile (transposed) while P is being written/consumed
    if (have_next) {
#pragma unroll
      for (int it = 0; it < 4; ++it) {
        union { uint4 q; unsigned short s[8]; } vu;
        vu.q = vreg[it];
#pragma unroll
        for (int e = 0; e < 8; ++e) Vt[nxt][sch + e][srow[it]] = vu.s[e];
      }
    }

    // O += P @ V  (per-wave Ps, same-wave LDS ordering is in-order)
#pragma unroll
    for (int dt = 0; dt < 4; ++dt)
#pragma unroll
      for (int ks = 0; ks < 2; ++ks) {
        const unsigned short* pa = &Ps[wave][lo][hi * 8 + ks * 32];
        v16bf af = ld_frag(pa, pa + 16);
        const unsigned short* pb = &Vt[cur][dt * 16 + lo][hi * 8 + ks * 32];
        v16bf bfr = ld_frag(pb, pb + 16);
        accO[dt] = __builtin_amdgcn_wmma_f32_16x16x32_bf16(
            false, af, false, bfr, (short)0, accO[dt], false, false);
      }

    if (have_next) async_wait0();
    __syncthreads();
  }

#pragma unroll
  for (int dt = 0; dt < 4; ++dt)
#pragma unroll
    for (int r = 0; r < 8; ++r) {
      int t = tbase + r + 8 * hi;
      int d = dt * 16 + lo;
      float v = accO[dt][r] / lrun[r];
      yb[(((size_t)b * 2048 + t) * 16 + h) * 64 + d] = f2bf(v);  // [b,t,h,d]
    }
}

// ---------------------------------------------------------------------------
extern "C" void kernel_launch(void* const* d_in, const int* in_sizes, int n_in,
                              void* d_out, int out_size, void* d_ws, size_t ws_size,
                              hipStream_t stream) {
  (void)in_sizes; (void)n_in; (void)out_size; (void)ws_size;
  const float* x    = (const float*)d_in[0];
  const float* cosT = (const float*)d_in[1];
  const float* sinT = (const float*)d_in[2];
  const float* Wq   = (const float*)d_in[3];
  const float* Wk   = (const float*)d_in[4];
  const float* Wv   = (const float*)d_in[5];
  const float* Wo   = (const float*)d_in[6];

  char* ws = (char*)d_ws;
  size_t off = 0;
  auto take = [&](size_t bytes) -> char* {
    char* p = ws + off;
    off = (off + bytes + 255) & ~(size_t)255;
    return p;
  };
  unsigned short* xb    = (unsigned short*)take((size_t)4096 * 1024 * 2);
  unsigned short* wqkvb = (unsigned short*)take((size_t)1024 * 1536 * 2);
  unsigned short* wob   = (unsigned short*)take((size_t)1024 * 1024 * 2);
  float*          qkv   = (float*)take((size_t)4096 * 1536 * 4);
  unsigned short* qbuf  = (unsigned short*)take((size_t)2 * 16 * 2048 * 64 * 2);
  unsigned short* kbuf  = (unsigned short*)take((size_t)2 * 4 * 2048 * 64 * 2);
  unsigned short* vbuf  = (unsigned short*)take((size_t)2 * 4 * 2048 * 64 * 2);
  unsigned short* ybuf  = (unsigned short*)take((size_t)4096 * 1024 * 2);

  cvt_f32_bf16<<<2048, 256, 0, stream>>>(x, xb, 4096 * 1024);
  cvt_f32_bf16<<<1024, 256, 0, stream>>>(Wo, wob, 1024 * 1024);
  pack_wqkv<<<1024, 256, 0, stream>>>(Wq, Wk, Wv, wqkvb);

  // QKV projection: [4096,1024] @ [1024,1536]
  gemm_bf16_f32<<<dim3(1536 / 128, 4096 / 128), 256, 0, stream>>>(
      xb, wqkvb, qkv, 4096, 1536, 1024);

  // RoPE + RMSNorm + layout to head-major bf16 (B*T*(H+2*KVH) waves)
  rope_rms<<<(2 * 2048 * 24) / 4, 128, 0, stream>>>(qkv, cosT, sinT, qbuf, kbuf, vbuf);

  // Flash attention: (T/64, B*H)
  attn_fwd<<<dim3(2048 / 64, 2 * 16), 128, 0, stream>>>(qbuf, kbuf, vbuf, ybuf);

  // Output projection: [4096,1024] @ [1024,1024] -> f32 out
  gemm_bf16_f32<<<dim3(1024 / 128, 4096 / 128), 256, 0, stream>>>(
      ybuf, wob, (float*)d_out, 4096, 1024, 1024);
}
